// LidarSeedEncoder_70841190580297
// MI455X (gfx1250) — compile-verified, matrix operand order checked by reading, not run on hardware
//
#include <hip/hip_runtime.h>
#include <hip/hip_bf16.h>

typedef __attribute__((ext_vector_type(16))) _Float16 v16h;
typedef __attribute__((ext_vector_type(8)))  _Float16 v8h;
typedef __attribute__((ext_vector_type(8)))  float    v8f;

namespace cfg {
constexpr int   B = 8;
constexpr int   N = 200000;
constexpr int   NX = 512, NY = 512;
constexpr int   NCELLS = NX * NY;   // 262144 = 2^18
constexpr int   Q = 500;
constexpr int   D = 256;
constexpr float X_MIN = -51.2f, X_MAX = 51.2f;
constexpr float Y_MIN = -51.2f, Y_MAX = 51.2f;
constexpr float Z_MIN = -5.0f,  Z_MAX = 3.0f;
constexpr float SIZE_X = 0.2f,  SIZE_Y = 0.2f;
constexpr int   TILE = 2048;        // LDS sort tile (16 KB of u64)
constexpr int   KEEP = 512;         // survivors per sorted tile (>= Q)
}

// ---------------------------------------------------------------- zero sums
__global__ void k_zero(float4* p, int n4) {
    int i = blockIdx.x * blockDim.x + threadIdx.x;
    if (i < n4) p[i] = make_float4(0.f, 0.f, 0.f, 0.f);
}

// ---------------------------------------------------------------- scatter points -> per-cell sums [B][NCELLS][5]
__global__ void k_scatter(const float4* __restrict__ pts,
                          const unsigned char* __restrict__ mask,
                          float* __restrict__ sums) {
    int id = blockIdx.x * blockDim.x + threadIdx.x;
    if (id >= cfg::B * cfg::N) return;
    float4 p = pts[id];
    bool v = (mask[id] != 0) &&
             p.x >= cfg::X_MIN && p.x <= cfg::X_MAX &&
             p.y >= cfg::Y_MIN && p.y <= cfg::Y_MAX &&
             p.z >= cfg::Z_MIN && p.z <= cfg::Z_MAX;
    if (!v) return;
    int gx = (int)floorf((p.x - cfg::X_MIN) / cfg::SIZE_X);
    int gy = (int)floorf((p.y - cfg::Y_MIN) / cfg::SIZE_Y);
    gx = min(max(gx, 0), cfg::NX - 1);
    gy = min(max(gy, 0), cfg::NY - 1);
    int b = id / cfg::N;
    size_t base = ((size_t)b * cfg::NCELLS + (size_t)(gx * cfg::NY + gy)) * 5u;
    atomicAdd(sums + base + 0, p.x);
    atomicAdd(sums + base + 1, p.y);
    atomicAdd(sums + base + 2, p.z);
    atomicAdd(sums + base + 3, p.w);
    atomicAdd(sums + base + 4, 1.0f);
}

// ---------------------------------------------------------------- full descending bitonic sort of one 2048-key LDS tile
__device__ __forceinline__ void bitonic2048(unsigned long long* s, unsigned t) {
    for (unsigned k = 2; k <= (unsigned)cfg::TILE; k <<= 1) {
        for (unsigned j = k >> 1; j >= 1; j >>= 1) {
            unsigned il = ((t & ~(j - 1)) << 1) | (t & (j - 1)); // lower index of pair
            unsigned ih = il + j;
            bool up = ((il & k) == 0);
            unsigned long long a = s[il], b = s[ih];
            bool sw = up ? (a < b) : (a > b);   // flipped comparator -> descending
            if (sw) { s[il] = b; s[ih] = a; }
            __syncthreads();
        }
    }
}

// ---------------------------------------------------------------- selection stage 0: build keys from counts, sort tiles
// key = (count<<32) | (NCELLS-1-cell)  => descending key order == (count desc, cell asc), keys unique.
__global__ __launch_bounds__(1024)
void k_sort_first(const float* __restrict__ sums, unsigned long long* __restrict__ keys) {
    __shared__ unsigned long long s[cfg::TILE];
    const unsigned t = threadIdx.x;
    const unsigned cellBase = blockIdx.x * cfg::TILE;
    const size_t   batchCell = (size_t)blockIdx.y * cfg::NCELLS;
    #pragma unroll
    for (int e = 0; e < 2; ++e) {
        unsigned idx = cellBase + t + e * 1024;
        unsigned c = (unsigned)(sums[(batchCell + idx) * 5u + 4] + 0.5f);
        s[t + e * 1024] = ((unsigned long long)c << 32) | (unsigned)(cfg::NCELLS - 1 - idx);
    }
    __syncthreads();
    bitonic2048(s, t);
    unsigned long long* g = keys + batchCell + cellBase;
    g[t]        = s[t];
    g[t + 1024] = s[t + 1024];
}

// ---------------------------------------------------------------- selection merge: gather top-KEEP of 4 sorted tiles,
// sort the 2048 candidates, emit one sorted tile. Global top-Q survives every pruning (Q <= KEEP).
__global__ __launch_bounds__(1024)
void k_sort_merge(const unsigned long long* __restrict__ in, unsigned long long* __restrict__ out,
                  unsigned inStride, unsigned outStride, int nSrcTiles) {
    __shared__ unsigned long long s[cfg::TILE];
    const unsigned t = threadIdx.x;
    const unsigned long long* gin = in + (size_t)blockIdx.y * inStride;
    #pragma unroll
    for (int e = 0; e < 2; ++e) {
        unsigned idx   = t + e * 1024;          // 0..2047
        unsigned chunk = idx >> 9;              // which of 4 source tiles
        unsigned offi  = idx & (cfg::KEEP - 1); // 0..511 within that tile's head
        int srcTile = blockIdx.x * 4 + chunk;
        s[idx] = (srcTile < nSrcTiles) ? gin[(size_t)srcTile * cfg::TILE + offi] : 0ull;
    }
    __syncthreads();
    bitonic2048(s, t);
    unsigned long long* g = out + (size_t)blockIdx.y * outStride + (size_t)blockIdx.x * cfg::TILE;
    g[t]        = s[t];
    g[t + 1024] = s[t + 1024];
}

// ---------------------------------------------------------------- gather top-Q: refs, scores, pf, rowcnt
__global__ void k_gather(const unsigned long long* __restrict__ topkeys, // [B][TILE], sorted desc
                         const float* __restrict__ sums,
                         float* __restrict__ refs, float* __restrict__ scores,
                         float* __restrict__ pf, float* __restrict__ rowcnt) {
    int id = blockIdx.x * blockDim.x + threadIdx.x;     // b*Q + q
    if (id >= cfg::B * cfg::Q) return;
    int b = id / cfg::Q, q = id - b * cfg::Q;
    unsigned long long key = topkeys[(size_t)b * cfg::TILE + q];
    unsigned cell = (unsigned)(cfg::NCELLS - 1) - (unsigned)(key & 0xffffffffu);
    float cnt = (float)(unsigned)(key >> 32);
    const float* s = sums + ((size_t)b * cfg::NCELLS + cell) * 5u;
    float denom = fmaxf(cnt, 1.0f);
    float fx = s[0] / denom, fy = s[1] / denom, fz = s[2] / denom, fi = s[3] / denom;
    bool valid = cnt > 0.0f;
    refs[id * 3 + 0] = valid ? fx : 0.0f;
    refs[id * 3 + 1] = valid ? fy : 0.0f;
    refs[id * 3 + 2] = valid ? fz : 0.0f;
    scores[id]       = valid ? cnt : 0.0f;
    pf[id * 5 + 0] = fx; pf[id * 5 + 1] = fy; pf[id * 5 + 2] = fz;
    pf[id * 5 + 3] = fi; pf[id * 5 + 4] = cnt;
    rowcnt[id] = cnt;
}

// ---------------------------------------------------------------- layer 1: h = relu(pf @ W1 + b1), write f16
__global__ void k_layer1(const float* __restrict__ pf, const float* __restrict__ W1,
                         const float* __restrict__ b1, _Float16* __restrict__ hf) {
    int id = blockIdx.x * blockDim.x + threadIdx.x;     // row*D + col
    if (id >= cfg::B * cfg::Q * cfg::D) return;
    int row = id >> 8, col = id & 255;
    const float* p = pf + row * 5;
    float acc = b1[col];
    acc = fmaf(p[0], W1[0 * cfg::D + col], acc);
    acc = fmaf(p[1], W1[1 * cfg::D + col], acc);
    acc = fmaf(p[2], W1[2 * cfg::D + col], acc);
    acc = fmaf(p[3], W1[3 * cfg::D + col], acc);
    acc = fmaf(p[4], W1[4 * cfg::D + col], acc);
    hf[id] = (_Float16)fmaxf(acc, 0.0f);
}

// ---------------------------------------------------------------- transpose+convert W2 -> W2T f16 (n-major)
__global__ void k_w2t(const float* __restrict__ W2, _Float16* __restrict__ w2t) {
    int id = blockIdx.x * blockDim.x + threadIdx.x;     // k*D + n
    if (id >= cfg::D * cfg::D) return;
    int k = id >> 8, n = id & 255;
    w2t[n * cfg::D + k] = (_Float16)W2[id];
}

// ---------------------------------------------------------------- layer 2 GEMM via v_wmma_f32_16x16x32_f16
// Block = 128 threads (4 waves). Each block: one 16-row tile. Each wave: 4 n-tiles (64 cols).
// A (h) row-major f16; B = W2T (n,k) f16 so per-lane fragments are two contiguous 16B loads.
// W2T has zero intra-block reuse (one full pass per block), so direct global loads +
// L2 residency (128 KB << 192 MB) beat LDS staging.
__global__ __launch_bounds__(128) void k_gemm2(const _Float16* __restrict__ A,
                                               const _Float16* __restrict__ Bt,
                                               const float* __restrict__ b2,
                                               const float* __restrict__ rowcnt,
                                               float* __restrict__ out) {
    const int rowtile = blockIdx.x;                 // 0..249  (4000 rows)
    const int wave = threadIdx.x >> 5;              // 0..3
    const int lane = threadIdx.x & 31;
    const int m    = lane & 15;
    const int hl   = lane >> 4;                     // half-wave selector
    const int ntb  = wave * 4;                      // base n-tile

    v8f acc[4] = {v8f{}, v8f{}, v8f{}, v8f{}};

    for (int ks = 0; ks < 8; ++ks) {
        const int kbase = ks * 32;
        // A fragment: lane (m, hl) needs K in [kbase+hl*8, +8) and [kbase+16+hl*8, +8)
        const _Float16* arow = A + (size_t)(rowtile * 16 + m) * cfg::D + kbase + hl * 8;
        v8h alo = *(const v8h*)(arow);
        v8h ahi = *(const v8h*)(arow + 16);
        v16h af = __builtin_shufflevector(alo, ahi, 0,1,2,3,4,5,6,7,8,9,10,11,12,13,14,15);
        #pragma unroll
        for (int t = 0; t < 4; ++t) {
            const int n = (ntb + t) * 16 + m;       // lane's column
            const _Float16* brow = Bt + (size_t)n * cfg::D + kbase + hl * 8;
            v8h blo = *(const v8h*)(brow);
            v8h bhi = *(const v8h*)(brow + 16);
            v16h bf = __builtin_shufflevector(blo, bhi, 0,1,2,3,4,5,6,7,8,9,10,11,12,13,14,15);
            acc[t] = __builtin_amdgcn_wmma_f32_16x16x32_f16(
                false, af, false, bf, (short)0, acc[t], false, false);
        }
    }

    // per-lane validity for the 8 rows this lane's accumulators cover
    float cnt[8];
    #pragma unroll
    for (int r = 0; r < 8; ++r) cnt[r] = rowcnt[rowtile * 16 + r + 8 * hl];

    #pragma unroll
    for (int t = 0; t < 4; ++t) {
        const int col = (ntb + t) * 16 + m;
        const float bias = b2[col];
        #pragma unroll
        for (int r = 0; r < 8; ++r) {
            const int row = rowtile * 16 + r + 8 * hl;
            out[(size_t)row * cfg::D + col] = (cnt[r] > 0.0f) ? (acc[t][r] + bias) : 0.0f;
        }
    }
}

// ---------------------------------------------------------------- host
extern "C" void kernel_launch(void* const* d_in, const int* in_sizes, int n_in,
                              void* d_out, int out_size, void* d_ws, size_t ws_size,
                              hipStream_t stream) {
    (void)in_sizes; (void)n_in; (void)out_size; (void)ws_size;
    const float4*        points = (const float4*)d_in[0];
    const unsigned char* mask   = (const unsigned char*)d_in[1];
    const float*         W1     = (const float*)d_in[2];
    const float*         b1     = (const float*)d_in[3];
    const float*         W2     = (const float*)d_in[4];
    const float*         b2     = (const float*)d_in[5];
    float* out = (float*)d_out;

    float* queries = out;                                           // 8*500*256
    float* refs    = out + (size_t)cfg::B * cfg::Q * cfg::D;        // 8*500*3
    float* scores  = refs + (size_t)cfg::B * cfg::Q * 3;            // 8*500

    char* ws = (char*)d_ws;
    size_t off = 0;
    auto carve = [&](size_t bytes) -> void* {
        void* p = ws + off;
        off = (off + bytes + 255) & ~(size_t)255;
        return p;
    };
    typedef unsigned long long u64;
    float* sums  = (float*)carve((size_t)cfg::B * cfg::NCELLS * 5 * sizeof(float));
    u64*   keys  = (u64*)carve((size_t)cfg::B * cfg::NCELLS * 8);   // stage-0 sorted tiles
    u64*   cand1 = (u64*)carve((size_t)cfg::B * 65536 * 8);
    u64*   cand2 = (u64*)carve((size_t)cfg::B * 16384 * 8);
    u64*   cand3 = (u64*)carve((size_t)cfg::B * 4096 * 8);
    u64*   cand4 = (u64*)carve((size_t)cfg::B * cfg::TILE * 8);     // final sorted top candidates
    float*    pf     = (float*)carve((size_t)cfg::B * cfg::Q * 5 * sizeof(float));
    float*    rowcnt = (float*)carve((size_t)cfg::B * cfg::Q * sizeof(float));
    _Float16* hf     = (_Float16*)carve((size_t)cfg::B * cfg::Q * cfg::D * sizeof(_Float16));
    _Float16* w2t    = (_Float16*)carve((size_t)cfg::D * cfg::D * sizeof(_Float16));

    // 1) zero accumulator grid (B*NCELLS*5 floats, divisible by 4)
    {
        int n4 = cfg::B * cfg::NCELLS * 5 / 4;
        k_zero<<<(n4 + 255) / 256, 256, 0, stream>>>((float4*)sums, n4);
    }
    // 2) scatter points
    {
        int n = cfg::B * cfg::N;
        k_scatter<<<(n + 255) / 256, 256, 0, stream>>>(points, mask, sums);
    }
    // 3) top-Q selection: tile sort + prune tournament (exact; keys unique so order is total)
    //    262144 -> 65536 -> 16384 -> 4096 -> 1024(+pad) candidates, 5 launches.
    k_sort_first<<<dim3(cfg::NCELLS / cfg::TILE, cfg::B), 1024, 0, stream>>>(sums, keys);
    k_sort_merge<<<dim3(32, cfg::B), 1024, 0, stream>>>(keys,  cand1, cfg::NCELLS, 65536, 128);
    k_sort_merge<<<dim3(8,  cfg::B), 1024, 0, stream>>>(cand1, cand2, 65536, 16384, 32);
    k_sort_merge<<<dim3(2,  cfg::B), 1024, 0, stream>>>(cand2, cand3, 16384, 4096, 8);
    k_sort_merge<<<dim3(1,  cfg::B), 1024, 0, stream>>>(cand3, cand4, 4096, cfg::TILE, 2);
    // 4) gather top-Q
    {
        int n = cfg::B * cfg::Q;
        k_gather<<<(n + 255) / 256, 256, 0, stream>>>(cand4, sums, refs, scores, pf, rowcnt);
    }
    // 5) layer 1 (K=5, scalar) -> h f16
    {
        int n = cfg::B * cfg::Q * cfg::D;
        k_layer1<<<(n + 255) / 256, 256, 0, stream>>>(pf, W1, b1, hf);
    }
    // 6) W2 -> transposed f16
    {
        int n = cfg::D * cfg::D;
        k_w2t<<<(n + 255) / 256, 256, 0, stream>>>(W2, w2t);
    }
    // 7) layer 2 GEMM via WMMA + bias + slot-valid mask
    k_gemm2<<<cfg::B * cfg::Q / 16, 128, 0, stream>>>(hf, w2t, b2, rowcnt, queries);
}
